// CIN_4432406249709
// MI455X (gfx1250) — matrix operand order repaired and checked by reading, not compiled, not run on hardware
//
#include <hip/hip_runtime.h>

// ---- plain ext-vector types (trivially constructible -> union-safe) ----
typedef __attribute__((ext_vector_type(4)))  float          v4f;
typedef __attribute__((ext_vector_type(4)))  unsigned int   v4u;
typedef __attribute__((ext_vector_type(8)))  float          v8f;
typedef __attribute__((ext_vector_type(16))) __bf16         v16bf;

union BfVec {
    v16bf b;    // feed to WMMA builtin / element-wise build
    v4u   q[2]; // 2x16B loads/stores
};

// split f32 into bf16 hi + bf16 lo (hardware RNE converts)
static __device__ __forceinline__ void split_bf16(float x, __bf16& h, __bf16& l) {
    h = (__bf16)x;
    l = (__bf16)(x - (float)h);
}

// -------------------------------------------------------------------------
// prep: W [256][4096] f32 -> Whi/Wlo bf16 in A-fragment order.
// Fragment element e of lane (row o, half hig) within 32-col chunk:
//   K = (e>>3)*16 + hig*8 + (e&7)
// Packed address: ((o*128 + chunk)*2 + hig)*16 elements, contiguous 32B.
// -------------------------------------------------------------------------
__global__ __launch_bounds__(256) void prep_w_kernel(
    const float* __restrict__ W,
    unsigned short* __restrict__ Whi, unsigned short* __restrict__ Wlo)
{
    const int idx   = blockIdx.x * 256 + threadIdx.x;  // 0..65535
    const int hig   = idx & 1;
    const int chunk = (idx >> 1) & 127;
    const int o     = idx >> 8;

    const float* src = W + (size_t)o * 4096 + chunk * 32 + hig * 8;
    v4f f0 = *(const v4f*)(src);
    v4f f1 = *(const v4f*)(src + 4);
    v4f f2 = *(const v4f*)(src + 16);
    v4f f3 = *(const v4f*)(src + 20);
    float fv[16] = { f0.x, f0.y, f0.z, f0.w,  f1.x, f1.y, f1.z, f1.w,
                     f2.x, f2.y, f2.z, f2.w,  f3.x, f3.y, f3.z, f3.w };

    BfVec H, L;
    #pragma unroll
    for (int e = 0; e < 16; ++e) {
        __bf16 h, l;
        split_bf16(fv[e], h, l);
        H.b[e] = h; L.b[e] = l;
    }
    v4u* dh = (v4u*)(Whi + (size_t)idx * 16);
    v4u* dl = (v4u*)(Wlo + (size_t)idx * 16);
    dh[0] = H.q[0]; dh[1] = H.q[1];
    dl[0] = L.q[0]; dl[1] = L.q[1];
}

// -------------------------------------------------------------------------
// out[b,o,k] = sum_{h,m} W[o,h*64+m] * xl[b,h,k] * x0[b,m,k] + bias[o]
// Grid: 512 blocks = (batch b) x (k-half). Block: 256 threads = 8 waves.
// Wave w owns OUT rows [32w,32w+32) x 64 k columns -> 2x4 tiles of 16x16.
// PACKED: A fragments come pre-split/pre-swizzled from ws (no VALU).
// -------------------------------------------------------------------------
template <bool PACKED>
__global__ __launch_bounds__(256) void cin_wmma_kernel(
    const float* __restrict__ x0, const float* __restrict__ xl,
    const float* __restrict__ Wraw,
    const unsigned short* __restrict__ Whi, const unsigned short* __restrict__ Wlo,
    const float* __restrict__ bias, float* __restrict__ out)
{
    // P row stride = 72 bf16 (144B) -> every b128 access stays 16B aligned.
    __shared__ float  xls[64 * 64];        // 16 KB
    __shared__ __bf16 Phi[2][64 * 72];     // 18 KB
    __shared__ __bf16 Plo[2][64 * 72];     // 18 KB

    const int b     = blockIdx.x >> 1;
    const int kbase = (blockIdx.x & 1) * 64;
    const int t     = threadIdx.x;
    const int wave  = t >> 5;
    const int lane  = t & 31;
    const int r     = lane & 15;   // N column / A row within tile
    const int hig   = lane >> 4;   // half-wave select

    // ---- stage xl[b, :, kbase:kbase+64] into LDS (coalesced f32x4) ----
    {
        const int h  = t >> 2;
        const int kk = (t & 3) * 16;
        const v4f* src = (const v4f*)(xl + ((size_t)b * 64 + h) * 128 + kbase + kk);
        v4f* dst = (v4f*)(xls + h * 64 + kk);
        dst[0] = src[0]; dst[1] = src[1]; dst[2] = src[2]; dst[3] = src[3];
    }

    // ---- each thread keeps one x0 column slice in registers ----
    const int kloc = t >> 2;          // 0..63 : local k column
    const int mg   = (t & 3) * 16;    // 16 m values per thread
    float x0r[16];
    {
        const float* src = x0 + ((size_t)b * 64 + mg) * 128 + kbase + kloc;
        #pragma unroll
        for (int j = 0; j < 16; ++j) x0r[j] = src[(size_t)j * 128];
    }

    // ---- accumulators, initialized with bias (C starts at bias) ----
    v8f acc[2][4];
    #pragma unroll
    for (int tt = 0; tt < 2; ++tt) {
        const int o0 = wave * 32 + tt * 16 + hig * 8;   // 32B aligned
        v4f b0 = *(const v4f*)(bias + o0);
        v4f b1 = *(const v4f*)(bias + o0 + 4);
        v8f c;
        c[0] = b0.x; c[1] = b0.y; c[2] = b0.z; c[3] = b0.w;
        c[4] = b1.x; c[5] = b1.y; c[6] = b1.z; c[7] = b1.w;
        #pragma unroll
        for (int n = 0; n < 4; ++n) acc[tt][n] = c;
    }

    // ---- build P[k][m] = x0[m,k]*xl[h,k], split hi/lo bf16, into LDS ----
    auto buildP = [&](int h, int buf) {
        const float xlv = xls[h * 64 + kloc];
        BfVec H, L;
        #pragma unroll
        for (int j = 0; j < 16; ++j) {
            __bf16 ph, pl;
            split_bf16(x0r[j] * xlv, ph, pl);
            H.b[j] = ph; L.b[j] = pl;
        }
        v4u* dh = (v4u*)&Phi[buf][kloc * 72 + mg];
        v4u* dl = (v4u*)&Plo[buf][kloc * 72 + mg];
        dh[0] = H.q[0]; dh[1] = H.q[1];
        dl[0] = L.q[0]; dl[1] = L.q[1];
    };

    // ---- WMMA accumulation for one h ----
    auto doWmma = [&](int h, int buf) {
        #pragma unroll
        for (int mc = 0; mc < 2; ++mc) {     // 32-deep K chunks of the m-dim
            BfVec Ahi[2], Alo[2];
            #pragma unroll
            for (int tt = 0; tt < 2; ++tt) {
                const int o = wave * 32 + tt * 16 + r;
                if constexpr (PACKED) {
                    const size_t base = (((size_t)o * 128 + (h * 2 + mc)) * 2 + hig) * 16;
                    const v4u* ph = (const v4u*)(Whi + base);
                    const v4u* pl = (const v4u*)(Wlo + base);
                    Ahi[tt].q[0] = ph[0]; Ahi[tt].q[1] = ph[1];
                    Alo[tt].q[0] = pl[0]; Alo[tt].q[1] = pl[1];
                } else {
                    const float* wr = Wraw + (size_t)o * 4096 + h * 64 + mc * 32 + hig * 8;
                    v4f f0 = *(const v4f*)(wr);
                    v4f f1 = *(const v4f*)(wr + 4);
                    v4f f2 = *(const v4f*)(wr + 16);
                    v4f f3 = *(const v4f*)(wr + 20);
                    float fv[16] = { f0.x, f0.y, f0.z, f0.w,  f1.x, f1.y, f1.z, f1.w,
                                     f2.x, f2.y, f2.z, f2.w,  f3.x, f3.y, f3.z, f3.w };
                    #pragma unroll
                    for (int e = 0; e < 16; ++e) {
                        __bf16 h2, l2;
                        split_bf16(fv[e], h2, l2);
                        Ahi[tt].b[e] = h2; Alo[tt].b[e] = l2;
                    }
                }
            }
            #pragma unroll
            for (int n = 0; n < 4; ++n) {
                // B 32x16 bf16 layout: elem e -> K = (e>>3)*16 + hig*8 + (e&7), col N = r
                const int row = (n * 16 + r) * 72 + mc * 32 + hig * 8;
                BfVec Bhi, Blo;
                const v4u* ph = (const v4u*)&Phi[buf][row];
                const v4u* pl = (const v4u*)&Plo[buf][row];
                Bhi.q[0] = ph[0]; Bhi.q[1] = ph[2];
                Blo.q[0] = pl[0]; Blo.q[1] = pl[2];
                #pragma unroll
                for (int tt = 0; tt < 2; ++tt) {
                    acc[tt][n] = __builtin_amdgcn_wmma_f32_16x16x32_bf16(
                        false, Ahi[tt].b, false, Bhi.b, (short)0, acc[tt][n], false, false);
                    acc[tt][n] = __builtin_amdgcn_wmma_f32_16x16x32_bf16(
                        false, Alo[tt].b, false, Bhi.b, (short)0, acc[tt][n], false, false);
                    acc[tt][n] = __builtin_amdgcn_wmma_f32_16x16x32_bf16(
                        false, Ahi[tt].b, false, Blo.b, (short)0, acc[tt][n], false, false);
                }
            }
        }
    };

    // ---- software-pipelined h loop: build P(h+1) while WMMA consumes P(h) ----
    buildP(0, 0);
    __syncthreads();
    #pragma unroll 1
    for (int h = 0; h < 64; ++h) {
        const int buf = h & 1;
        if (h < 63) buildP(h + 1, buf ^ 1);
        doWmma(h, buf);
        __syncthreads();
    }

    // ---- store: C/D layout M = hig*8 + i, N = r ----
    #pragma unroll
    for (int tt = 0; tt < 2; ++tt) {
        #pragma unroll
        for (int n = 0; n < 4; ++n) {
            const int kcol = kbase + n * 16 + r;
            #pragma unroll
            for (int i = 0; i < 8; ++i) {
                const int o = wave * 32 + tt * 16 + hig * 8 + i;
                out[((size_t)b * 256 + o) * 128 + kcol] = acc[tt][n][i];
            }
        }
    }
}

extern "C" void kernel_launch(void* const* d_in, const int* in_sizes, int n_in,
                              void* d_out, int out_size, void* d_ws, size_t ws_size,
                              hipStream_t stream) {
    // setup_inputs order: x0 (f32), xl (f32), k (int scalar, unused), W (f32), b (f32)
    const float* x0   = (const float*)d_in[0];
    const float* xl   = (const float*)d_in[1];
    const float* W    = (const float*)d_in[3];
    const float* bias = (const float*)d_in[4];
    float* out = (float*)d_out;

    const size_t n_w  = (size_t)256 * 4096;           // W elements
    const size_t need = n_w * 2 * sizeof(unsigned short); // Whi + Wlo, 4 MB

    dim3 grid(512);   // 256 batches x 2 k-halves
    dim3 block(256);  // 8 waves of 32

    if (d_ws != nullptr && ws_size >= need) {
        unsigned short* Whi = (unsigned short*)d_ws;
        unsigned short* Wlo = Whi + n_w;
        prep_w_kernel<<<dim3(256), block, 0, stream>>>(W, Whi, Wlo);
        cin_wmma_kernel<true><<<grid, block, 0, stream>>>(x0, xl, W, Whi, Wlo, bias, out);
    } else {
        cin_wmma_kernel<false><<<grid, block, 0, stream>>>(x0, xl, W, nullptr, nullptr, bias, out);
    }
}
